// MultiheadSelfAttention_46471546143466
// MI455X (gfx1250) — matrix operand order repaired and checked by reading, not compiled, not run on hardware
//
#include <hip/hip_runtime.h>
#include <hip/hip_bf16.h>
#include <math.h>

// ---------------------------------------------------------------------------
// MI455X (gfx1250) multi-head self-attention with RoPE.
// B=2, S=2048, D=1024, H=16, hd=64. wave32, WMMA f16->f32,
// async global->LDS staging for the shared GEMM A-tile.
// Single base-pointer-per-stream addressing so subtile/k terms fold into
// the 24-bit global-load immediate offsets (kills the round-2 scratch spill).
// ---------------------------------------------------------------------------

typedef _Float16 half_t;
typedef __attribute__((ext_vector_type(16))) _Float16 v16h;
typedef __attribute__((ext_vector_type(8)))  _Float16 v8h;
typedef __attribute__((ext_vector_type(8)))  float    v8f;

#define BD    2
#define SD    2048
#define DD    1024
#define HD    16
#define HDIM  64
#define NTOK  (BD * SD)          // 4096 tokens
#define NEG_BIG (-3.0e38f)
#define LOG_THETA 9.210340371976184f   // ln(10000)

static __device__ __forceinline__ v8f wmma_f16(v16h a, v16h b, v8f c) {
  return __builtin_amdgcn_wmma_f32_16x16x32_f16(
      false, a, false, b, (short)0, c, false, false);
}

// A-matrix fragment (16-bit, 16x32): lane half hl holds K {c..c+15},
// c = hl*16.  p points at element (row, kbase + c).  Works for global or LDS.
static __device__ __forceinline__ v16h load_afrag(const half_t* p) {
  v8h lo = *(const v8h*)(p);
  v8h hi = *(const v8h*)(p + 8);
  v16h r;
#pragma unroll
  for (int i = 0; i < 8; ++i) { r[i] = lo[i]; r[i + 8] = hi[i]; }
  return r;
}

static __device__ __forceinline__ float hmax16(float v) {
#pragma unroll
  for (int m = 1; m <= 8; m <<= 1) v = fmaxf(v, __shfl_xor(v, m, 32));
  return v;   // reduce across each 16-lane half (C-layout rows)
}
static __device__ __forceinline__ float hsum16(float v) {
#pragma unroll
  for (int m = 1; m <= 8; m <<= 1) v += __shfl_xor(v, m, 32);
  return v;
}

// gfx1250 async global->LDS copy (16B), tracked by ASYNCcnt.
static __device__ __forceinline__ void async_copy_b128(unsigned lds_off,
                                                       const half_t* gptr) {
  unsigned long long ga = (unsigned long long)(size_t)gptr;
  asm volatile("global_load_async_to_lds_b128 %0, %1, off"
               :: "v"(lds_off), "v"(ga) : "memory");
}

// ---------------------------------------------------------------------------
// Kernel 0: fp32 -> f16 conversion
// ---------------------------------------------------------------------------
__global__ void conv_f32_f16(const float* __restrict__ src,
                             half_t* __restrict__ dst, int n) {
  int i = blockIdx.x * blockDim.x + threadIdx.x;
  if (i < n) dst[i] = (half_t)src[i];
}

// ---------------------------------------------------------------------------
// Kernel 1: fused QKV projection (C = x @ W^T) + RoPE.
// Block = 8 waves sharing a 32-token A tile staged through LDS with async
// copies (128-wide K chunks, barriers per chunk, uniform trip counts).
// Each wave computes a 32(M) x 64(N) tile = 8 WMMA accumulators.
//   z=0: Q -> Qh [b,h,s,hd]   z=1: K -> Kh [b,h,s,hd]   z=2: V -> Vt [b,h,hd,s]
// ---------------------------------------------------------------------------
#define APAD 136   // 128 + 8 halves row pad: 272B stride -> conflict-free b128
__global__ void qkv_rope_kernel(const half_t* __restrict__ xh,
                                const half_t* __restrict__ wq,
                                const half_t* __restrict__ wk,
                                const half_t* __restrict__ wv,
                                const int*    __restrict__ tp,
                                half_t* __restrict__ Qh,
                                half_t* __restrict__ Kh,
                                half_t* __restrict__ Vt) {
  __shared__ __align__(16) half_t abuf[32][APAD];

  const int tid  = threadIdx.x;
  const int lane = tid & 31;
  const int wave = tid >> 5;
  const int ln   = lane & 15;
  const int hl   = lane >> 4;
  const int m0   = blockIdx.x * 32;                 // token tile base
  const int n0   = blockIdx.y * 512 + wave * 64;    // out-dim tile base
  const int z    = blockIdx.z;                      // 0=Q 1=K 2=V
  const half_t* W = (z == 0) ? wq : (z == 1) ? wk : wv;

  // Single B base; t/k terms become immediate offsets (t*32KiB + kt*2B).
  const half_t* wbase = W + (size_t)(n0 + ln) * DD + (hl ? 16 : 0);
  // Single async-stage source base per thread.
  const int srow = tid >> 4;              // 0..15 (rows 0..15 for i=0)
  const int scol = (tid & 15) * 8;
  const half_t* xbase = xh + (size_t)(m0 + srow) * DD + scol;
  const unsigned lds0 = (unsigned)(size_t)&abuf[srow][scol];

  v8f acc[2][4];
#pragma unroll
  for (int mt = 0; mt < 2; ++mt)
#pragma unroll
    for (int t = 0; t < 4; ++t)
#pragma unroll
      for (int i = 0; i < 8; ++i) acc[mt][t][i] = 0.0f;

  for (int ck = 0; ck < DD / 128; ++ck) {
    // ---- async-stage x[m0..m0+31][ck*128 .. +127] into LDS --------------
    // thread handles (srow, scol) and (srow+16, scol)
    async_copy_b128(lds0,                      xbase + ck * 128);
    async_copy_b128(lds0 + 16 * APAD * 2,      xbase + ck * 128 + 16 * DD);
    asm volatile("s_wait_asynccnt 0" ::: "memory");
    __syncthreads();

#pragma unroll
    for (int kl = 0; kl < 128; kl += 32) {
      const int kt = ck * 128 + kl;
      v16h a[2];
#pragma unroll
      for (int mt = 0; mt < 2; ++mt)
        a[mt] = load_afrag(&abuf[mt * 16 + ln][kl + (hl ? 16 : 0)]);
#pragma unroll
      for (int t = 0; t < 4; ++t) {
        // B[k][n] = W[n][k]: lane = column n, 16 contiguous f16 along k
        v16h b = *(const v16h*)(wbase + t * 16 * DD + kt);
        acc[0][t] = wmma_f16(a[0], b, acc[0][t]);
        acc[1][t] = wmma_f16(a[1], b, acc[1][t]);
      }
    }
    asm volatile("s_wait_dscnt 0" ::: "memory");   // reads done before re-stage
    __syncthreads();
  }

  // Epilogue: C layout — lane half hl owns rows m = j + hl*8, column n = ln.
#pragma unroll
  for (int mt = 0; mt < 2; ++mt)
#pragma unroll
    for (int t = 0; t < 4; ++t) {
      const int o    = n0 + t * 16 + ln;   // absolute output dim
      const int head = o >> 6;
      const int d    = o & 63;
      const float inv_freq =
          __expf(-(float)(d & ~1) * (LOG_THETA / (float)HDIM));
#pragma unroll
      for (int j = 0; j < 8; ++j) {
        const int rowg = m0 + mt * 16 + j + hl * 8;   // global token index
        float c = acc[mt][t][j];
        float partner = __shfl_xor(c, 1, 32);  // RoPE pair = adjacent lane
        if (z < 2) {
          const int pos = tp[rowg];
          const float ang = (float)pos * inv_freq;
          const float cs = __cosf(ang), sn = __sinf(ang);
          c = (o & 1) ? (c * cs + partner * sn)
                      : (c * cs - partner * sn);
        }
        const int b  = rowg >> 11;
        const int s  = rowg & 2047;
        const int bh = b * HD + head;
        if (z == 0)
          Qh[((size_t)bh * SD + s) * HDIM + d] = (half_t)c;
        else if (z == 1)
          Kh[((size_t)bh * SD + s) * HDIM + d] = (half_t)c;
        else
          Vt[((size_t)bh * HDIM + d) * SD + s] = (half_t)c;
      }
    }
}

// ---------------------------------------------------------------------------
// Kernel 2: causal flash attention per (b,h).  One wave owns TWO 16-query
// tiles (32 queries) so each K/V fragment feeds two WMMAs.  Online softmax,
// P crosses C-layout -> A-layout via wave-private LDS (DS in-order per wave,
// no block barriers: causal trip counts diverge between waves).
// ---------------------------------------------------------------------------
__global__ void attn_kernel(const half_t* __restrict__ Qh,
                            const half_t* __restrict__ Kh,
                            const half_t* __restrict__ Vt,
                            half_t* __restrict__ Ah) {
  __shared__ __align__(32) _Float16 pst[8][2][16 * 32];  // per-wave P staging

  const int lane = threadIdx.x & 31;
  const int wave = threadIdx.x >> 5;
  const int ln   = lane & 15;
  const int hl   = lane >> 4;
  const int qw = blockIdx.x * 8 + wave;   // 32-query group (0..63)
  const int q0 = qw * 32;
  const int bh = blockIdx.y;              // 0..31
  const int b  = bh >> 4, h = bh & 15;

  // Single base per stream; tile/t terms fold into immediate offsets.
  const half_t* kbase =
      Kh + (size_t)bh * SD * HDIM + (size_t)ln * HDIM + (hl ? 16 : 0);
  const half_t* vbase =
      Vt + (size_t)bh * HDIM * SD + (size_t)ln * SD + (hl ? 16 : 0);

  v16h qf[2][2];
  {
    const half_t* qbase = Qh + (size_t)bh * SD * HDIM +
                          (size_t)(q0 + ln) * HDIM + (hl ? 16 : 0);
#pragma unroll
    for (int qi = 0; qi < 2; ++qi) {
      qf[qi][0] = load_afrag(qbase + qi * 16 * HDIM);
      qf[qi][1] = load_afrag(qbase + qi * 16 * HDIM + 32);
    }
  }

  float mrow[2][8], lrow[2][8];
  v8f o[2][4];
#pragma unroll
  for (int qi = 0; qi < 2; ++qi) {
#pragma unroll
    for (int j = 0; j < 8; ++j) { mrow[qi][j] = NEG_BIG; lrow[qi][j] = 0.0f; }
#pragma unroll
    for (int t = 0; t < 4; ++t)
#pragma unroll
      for (int i = 0; i < 8; ++i) o[qi][t][i] = 0.0f;
  }

  const int kend = (qw + 1) * 32;   // covers keys [0, q0+32) causally
  for (int kb = 0; kb < kend; kb += 32) {
    // ---- K fragments for this 32-key block (shared by both q tiles) ----
    v16h kf[2][2];
#pragma unroll
    for (int tile = 0; tile < 2; ++tile) {
      const half_t* krow = kbase + (size_t)(kb + tile * 16) * HDIM;
      kf[tile][0] = *(const v16h*)(krow);        // hd 0..31
      kf[tile][1] = *(const v16h*)(krow + 32);   // hd 32..63
    }

#pragma unroll
    for (int qi = 0; qi < 2; ++qi) {
      v8f s[2];
#pragma unroll
      for (int tile = 0; tile < 2; ++tile) {
#pragma unroll
        for (int i = 0; i < 8; ++i) s[tile][i] = 0.0f;
        s[tile] = wmma_f16(qf[qi][0], kf[tile][0], s[tile]);
        s[tile] = wmma_f16(qf[qi][1], kf[tile][1], s[tile]);
      }
      // ---- online softmax (C layout: row m = j + hl*8, col n = ln) -----
#pragma unroll
      for (int j = 0; j < 8; ++j) {
        const int query = q0 + qi * 16 + j + hl * 8;
        float a0 = (kb + ln      <= query) ? s[0][j] * 0.125f : NEG_BIG;
        float a1 = (kb + 16 + ln <= query) ? s[1][j] * 0.125f : NEG_BIG;
        const float bm    = hmax16(fmaxf(a0, a1));
        const float mnew  = fmaxf(mrow[qi][j], bm);
        const float alpha = __expf(mrow[qi][j] - mnew);
        mrow[qi][j] = mnew;
        const float p0 = __expf(a0 - mnew);
        const float p1 = __expf(a1 - mnew);
        lrow[qi][j] = alpha * lrow[qi][j] + hsum16(p0 + p1);
#pragma unroll
        for (int t = 0; t < 4; ++t) o[qi][t][j] *= alpha;
        const int m = j + hl * 8;
        pst[wave][qi][m * 32 + ln]      = (half_t)p0;
        pst[wave][qi][m * 32 + 16 + ln] = (half_t)p1;
      }
    }

    asm volatile("s_wait_dscnt 0" ::: "memory");  // wave-private staging done

    // ---- O += P(16x32) x V(32x64); Vt rows are key-contiguous ----------
    v16h vf[4];
#pragma unroll
    for (int t = 0; t < 4; ++t)
      vf[t] = *(const v16h*)(vbase + (size_t)t * 16 * SD + kb);
#pragma unroll
    for (int qi = 0; qi < 2; ++qi) {
      v16h pf;
      const _Float16* pr = &pst[wave][qi][ln * 32 + (hl ? 16 : 0)];
      v8h lo = *(const v8h*)(pr);
      v8h hi = *(const v8h*)(pr + 8);
#pragma unroll
      for (int i = 0; i < 8; ++i) { pf[i] = lo[i]; pf[i + 8] = hi[i]; }
#pragma unroll
      for (int t = 0; t < 4; ++t) o[qi][t] = wmma_f16(pf, vf[t], o[qi][t]);
    }
  }

  // ---- normalize and write att output [token][D] f16 -------------------
#pragma unroll
  for (int qi = 0; qi < 2; ++qi)
#pragma unroll
    for (int j = 0; j < 8; ++j) {
      const float inv = 1.0f / lrow[qi][j];
      const int s = q0 + qi * 16 + j + hl * 8;
#pragma unroll
      for (int t = 0; t < 4; ++t) {
        Ah[((size_t)b * SD + s) * DD + h * HDIM + t * 16 + ln] =
            (half_t)(o[qi][t][j] * inv);
      }
    }
}

// ---------------------------------------------------------------------------
// Kernel 3: output projection out = Ah @ Wo^T (f16 x f16 -> f32), 32x64/wave
// ---------------------------------------------------------------------------
__global__ void out_proj_kernel(const half_t* __restrict__ Ah,
                                const half_t* __restrict__ woh,
                                float* __restrict__ out) {
  const int lane = threadIdx.x & 31;
  const int wave = threadIdx.x >> 5;
  const int ln   = lane & 15;
  const int hl   = lane >> 4;
  const int m0 = blockIdx.x * 32;
  const int n0 = blockIdx.y * 512 + wave * 64;

  const half_t* abase = Ah + (size_t)(m0 + ln) * DD + (hl ? 16 : 0);
  const half_t* wbase = woh + (size_t)(n0 + ln) * DD + (hl ? 16 : 0);

  v8f acc[2][4];
#pragma unroll
  for (int mt = 0; mt < 2; ++mt)
#pragma unroll
    for (int t = 0; t < 4; ++t)
#pragma unroll
      for (int i = 0; i < 8; ++i) acc[mt][t][i] = 0.0f;

  for (int kt = 0; kt < DD; kt += 32) {
    v16h a[2];
#pragma unroll
    for (int mt = 0; mt < 2; ++mt)
      a[mt] = load_afrag(abase + mt * 16 * DD + kt);
#pragma unroll
    for (int t = 0; t < 4; ++t) {
      v16h bfr = *(const v16h*)(wbase + t * 16 * DD + kt);
      acc[0][t] = wmma_f16(a[0], bfr, acc[0][t]);
      acc[1][t] = wmma_f16(a[1], bfr, acc[1][t]);
    }
  }
#pragma unroll
  for (int mt = 0; mt < 2; ++mt)
#pragma unroll
    for (int t = 0; t < 4; ++t)
#pragma unroll
      for (int j = 0; j < 8; ++j) {
        const int m = mt * 16 + j + hl * 8;
        out[(size_t)(m0 + m) * DD + n0 + t * 16 + ln] = acc[mt][t][j];
      }
}

// ---------------------------------------------------------------------------
extern "C" void kernel_launch(void* const* d_in, const int* in_sizes, int n_in,
                              void* d_out, int out_size, void* d_ws,
                              size_t ws_size, hipStream_t stream) {
  const float* x  = (const float*)d_in[0];
  const int*   tp = (const int*)d_in[1];
  const float* Wq = (const float*)d_in[2];
  const float* Wk = (const float*)d_in[3];
  const float* Wv = (const float*)d_in[4];
  const float* Wo = (const float*)d_in[5];
  float* out = (float*)d_out;

  const size_t MB = (size_t)1 << 20;
  char* ws = (char*)d_ws;
  half_t* xh  = (half_t*)(ws + 0 * MB);    // 4096x1024    (8 MiB)
  half_t* wqh = (half_t*)(ws + 8 * MB);    // 1024x1024    (2 MiB)
  half_t* wkh = (half_t*)(ws + 10 * MB);
  half_t* wvh = (half_t*)(ws + 12 * MB);
  half_t* woh = (half_t*)(ws + 14 * MB);
  half_t* Qh  = (half_t*)(ws + 16 * MB);   // [b,h,s,hd]   (8 MiB)
  half_t* Kh  = (half_t*)(ws + 24 * MB);   // [b,h,s,hd]   (8 MiB)
  half_t* Vt  = (half_t*)(ws + 32 * MB);   // [b,h,hd,s]   (8 MiB)
  half_t* Ah  = (half_t*)(ws + 40 * MB);   // [tok, D]     (8 MiB)

  const int nx = NTOK * DD;
  const int nw = DD * DD;
  conv_f32_f16<<<(nx + 255) / 256, 256, 0, stream>>>(x, xh, nx);
  conv_f32_f16<<<(nw + 255) / 256, 256, 0, stream>>>(Wq, wqh, nw);
  conv_f32_f16<<<(nw + 255) / 256, 256, 0, stream>>>(Wk, wkh, nw);
  conv_f32_f16<<<(nw + 255) / 256, 256, 0, stream>>>(Wv, wvh, nw);
  conv_f32_f16<<<(nw + 255) / 256, 256, 0, stream>>>(Wo, woh, nw);

  // QKV + RoPE: 128 (32-token) M-tiles x 2 N-chunks x {Q,K,V}
  qkv_rope_kernel<<<dim3(NTOK / 32, 2, 3), 256, 0, stream>>>(
      xh, wqh, wkh, wvh, tp, Qh, Kh, Vt);

  // Flash attention: 8 (32-query) groups per block x 8 blocks, 32 (b,h)
  attn_kernel<<<dim3(SD / 32 / 8, BD * HD), 256, 0, stream>>>(Qh, Kh, Vt, Ah);

  // Output projection
  out_proj_kernel<<<dim3(NTOK / 32, 2), 256, 0, stream>>>(Ah, woh, out);
}
